// Attention_73658689126464
// MI455X (gfx1250) — matrix operand order repaired and checked by reading, not compiled, not run on hardware
//
#include <hip/hip_runtime.h>

typedef __attribute__((ext_vector_type(16))) __bf16 v16bf;
typedef __attribute__((ext_vector_type(8)))  __bf16 v8bf;
typedef __attribute__((ext_vector_type(8)))  float  v8f;
typedef __attribute__((ext_vector_type(4)))  float  f32x4;

namespace {
constexpr int kB = 8, kS = 1024, kH = 32, kKVH = 8, kD = 128;
constexpr int kG  = kH / kKVH;   // 4 query heads per kv head (one wave each)
constexpr int kQT = 16;          // query tile (WMMA M)
constexpr int kKT = 32;          // key tile (WMMA K for PV)
constexpr float kScale  = 0.08838834764831845f;  // 1/sqrt(128)
constexpr float kNegInf = -3.0e38f;
}

__device__ __forceinline__ unsigned short bf16_bits(float x) {
  __bf16 h = (__bf16)x;
  return __builtin_bit_cast(unsigned short, h);
}

__global__ __launch_bounds__(kG * 32) void paged_gqa_fa_kernel(
    const float* __restrict__ q, const float* __restrict__ k,
    const float* __restrict__ v, float* __restrict__ out) {
  __shared__ __bf16 kbuf[kKT * kD];        // [key][d], row-major bf16
  __shared__ __bf16 vtbuf[kD * kKT];       // [d][key], transposed bf16
  __shared__ __bf16 pbuf[kG * kQT * kKT];  // per-wave P scratch (C->A relayout)

  const int bid = blockIdx.x;
  const int qt  = bid & 63;          // S/16 = 64 query tiles
  const int kvh = (bid >> 6) & 7;    // KVH = 8
  const int b   = bid >> 9;          // B = 8

  const int tid  = threadIdx.x;
  const int wave = tid >> 5;
  const int lane = tid & 31;
  const int half = lane >> 4;        // which 16-lane half
  const int ln   = lane & 15;
  const int h    = kvh * kG + wave;  // query head = kvh*G + g

  // ---- Q tile (16 queries x 128 dims) -> A-operand registers, bf16 ----
  // A 16x32, contiguous k order used here (same as B): lane<16: m=lane, K elems {0..15}; lane>=16: m=lane-16,
  // K elems {16..31}. 4 chunks cover D=128.
  v16bf a_q[4];
  {
    const float* qrow = q + ((size_t)(b * kS + qt * kQT + ln) * kH + h) * kD;
    #pragma unroll
    for (int c = 0; c < 4; ++c) {
      const int off = c * 32 + half * 16;
      #pragma unroll
      for (int j = 0; j < 8; ++j) {
        a_q[c][j]     = (__bf16)qrow[off + j];
        a_q[c][8 + j] = (__bf16)qrow[off + 8 + j];
      }
    }
  }

  v8f o[8];                          // 16x128 output accumulator (8 N-tiles)
  #pragma unroll
  for (int t = 0; t < 8; ++t)
    #pragma unroll
    for (int i = 0; i < 8; ++i) o[t][i] = 0.0f;

  float mrow[8], lrow[8];            // per-row running max / sum
  #pragma unroll
  for (int r = 0; r < 8; ++r) { mrow[r] = kNegInf; lrow[r] = 0.0f; }

  const int qg_base = qt * kQT;
  const int n_kt = (qg_base + kQT - 1) / kKT + 1;  // causal tile bound

  for (int kt = 0; kt < n_kt; ++kt) {
    __syncthreads();  // previous iteration's LDS reads done (all 4 waves)

    // ---- K tile -> LDS (32 keys x 128 dims, bf16 row-major) ----
    {
      const int kk = tid >> 2;
      const int dc = (tid & 3) * 32;
      const float* krow =
          k + ((size_t)(b * kS + kt * kKT + kk) * kKVH + kvh) * kD + dc;
      unsigned pk[16];
      #pragma unroll
      for (int j = 0; j < 8; ++j) {
        f32x4 x = ((const f32x4*)krow)[j];
        pk[2 * j]     = (unsigned)bf16_bits(x[0]) | ((unsigned)bf16_bits(x[1]) << 16);
        pk[2 * j + 1] = (unsigned)bf16_bits(x[2]) | ((unsigned)bf16_bits(x[3]) << 16);
      }
      uint4* dst = (uint4*)&kbuf[kk * kD + dc];
      #pragma unroll
      for (int i = 0; i < 4; ++i)
        dst[i] = make_uint4(pk[4 * i], pk[4 * i + 1], pk[4 * i + 2], pk[4 * i + 3]);
    }
    // ---- V tile -> LDS transposed ([d][key]); coalesced global reads ----
    {
      const float* vcol =
          v + ((size_t)(b * kS + kt * kKT) * kKVH + kvh) * kD + tid;
      unsigned pk[16];
      #pragma unroll
      for (int j = 0; j < 16; ++j) {
        float x0 = vcol[(size_t)(2 * j) * kKVH * kD];
        float x1 = vcol[(size_t)(2 * j + 1) * kKVH * kD];
        pk[j] = (unsigned)bf16_bits(x0) | ((unsigned)bf16_bits(x1) << 16);
      }
      uint4* dst = (uint4*)&vtbuf[tid * kKT];
      #pragma unroll
      for (int i = 0; i < 4; ++i)
        dst[i] = make_uint4(pk[4 * i], pk[4 * i + 1], pk[4 * i + 2], pk[4 * i + 3]);
    }
    __syncthreads();

    // ---- S = Q K^T : two 16x16 score tiles (32 keys), K-dim 128 in 4 chunks ----
    v8f s0, s1;
    #pragma unroll
    for (int i = 0; i < 8; ++i) { s0[i] = 0.0f; s1[i] = 0.0f; }
    #pragma unroll
    for (int c = 0; c < 4; ++c) {
      // B, contiguous k order used here (same as A): lane<16: n=lane, K 0..15; lane>=16: n=lane-16, K 16..31
      v16bf bk0 = *(const v16bf*)&kbuf[(ln)      * kD + c * 32 + half * 16];
      v16bf bk1 = *(const v16bf*)&kbuf[(16 + ln) * kD + c * 32 + half * 16];
      s0 = __builtin_amdgcn_wmma_f32_16x16x32_bf16(false, a_q[c], false, bk0,
                                                   (short)0, s0, false, false);
      s1 = __builtin_amdgcn_wmma_f32_16x16x32_bf16(false, a_q[c], false, bk1,
                                                   (short)0, s1, false, false);
    }

    // ---- online softmax (C layout: lane<16 -> rows 0..7, lane>=16 -> 8..15) ----
    float p0[8], p1[8], alpha[8];
    #pragma unroll
    for (int r = 0; r < 8; ++r) {
      const int qg = qg_base + half * 8 + r;
      float x0 = s0[r] * kScale;
      float x1 = s1[r] * kScale;
      if (kt * kKT + ln      > qg) x0 = kNegInf;  // causal mask
      if (kt * kKT + 16 + ln > qg) x1 = kNegInf;
      float mx = fmaxf(x0, x1);
      mx = fmaxf(mx, __shfl_xor(mx, 1, 32));
      mx = fmaxf(mx, __shfl_xor(mx, 2, 32));
      mx = fmaxf(mx, __shfl_xor(mx, 4, 32));
      mx = fmaxf(mx, __shfl_xor(mx, 8, 32));
      const float nm = fmaxf(mrow[r], mx);
      alpha[r] = __expf(mrow[r] - nm);
      mrow[r]  = nm;
      p0[r] = __expf(x0 - nm);
      p1[r] = __expf(x1 - nm);
      float sum = p0[r] + p1[r];
      sum += __shfl_xor(sum, 1, 32);
      sum += __shfl_xor(sum, 2, 32);
      sum += __shfl_xor(sum, 4, 32);
      sum += __shfl_xor(sum, 8, 32);
      lrow[r] = lrow[r] * alpha[r] + sum;
    }
    #pragma unroll
    for (int t = 0; t < 8; ++t)
      #pragma unroll
      for (int r = 0; r < 8; ++r) o[t][r] *= alpha[r];

    // ---- P (f32 C-layout) -> LDS bf16 -> A-operand layout ----
    __bf16* pw = &pbuf[wave * (kQT * kKT)];
    #pragma unroll
    for (int r = 0; r < 8; ++r) {
      pw[(half * 8 + r) * kKT + ln]      = (__bf16)p0[r];
      pw[(half * 8 + r) * kKT + 16 + ln] = (__bf16)p1[r];
    }
    v16bf a_p;
    {
      const __bf16* prow = &pw[ln * kKT + half * 16];
      v8bf lo = *(const v8bf*)(prow);
      v8bf hi = *(const v8bf*)(prow + 8);
      a_p = __builtin_shufflevector(lo, hi, 0, 1, 2, 3, 4, 5, 6, 7,
                                    8, 9, 10, 11, 12, 13, 14, 15);
    }

    // ---- O += P V : 8 N-tiles of 16 dims each ----
    #pragma unroll
    for (int t = 0; t < 8; ++t) {
      v16bf bv = *(const v16bf*)&vtbuf[(t * 16 + ln) * kKT + half * 16];
      o[t] = __builtin_amdgcn_wmma_f32_16x16x32_bf16(false, a_p, false, bv,
                                                     (short)0, o[t], false, false);
    }
  }

  // ---- epilogue: normalize rows, store f32 ----
  #pragma unroll
  for (int r = 0; r < 8; ++r) {
    const float inv = 1.0f / lrow[r];
    float* orow =
        out + ((size_t)(b * kS + qt * kQT + half * 8 + r) * kH + h) * kD + ln;
    #pragma unroll
    for (int t = 0; t < 8; ++t) orow[t * 16] = o[t][r] * inv;
  }
}

extern "C" void kernel_launch(void* const* d_in, const int* in_sizes, int n_in,
                              void* d_out, int out_size, void* d_ws, size_t ws_size,
                              hipStream_t stream) {
  // inputs: q, k, v, k_cache, v_cache, slot_mapping, block_tables
  // The cache scatter+gather is an identity w.r.t. the output (slots written ==
  // slots read), and inputs must not be mutated, so attention reads q/k/v directly.
  const float* q = (const float*)d_in[0];
  const float* k = (const float*)d_in[1];
  const float* v = (const float*)d_in[2];
  float* out = (float*)d_out;
  dim3 grid(kB * kKVH * (kS / kQT));  // 8 * 8 * 64 = 4096 workgroups
  dim3 block(kG * 32);                // 4 waves = 4 query heads per kv head
  hipLaunchKernelGGL(paged_gqa_fa_kernel, grid, block, 0, stream, q, k, v, out);
  (void)in_sizes; (void)n_in; (void)out_size; (void)d_ws; (void)ws_size;
}
